// d_LSKblock5x3_1425929142557
// MI455X (gfx1250) — matrix operand order repaired and checked by reading, not compiled, not run on hardware
//
#include <hip/hip_runtime.h>
#include <hip/hip_bf16.h>

// ---------------------------------------------------------------------------
// Problem constants (match reference)
// ---------------------------------------------------------------------------
#define BATCH 4
#define CH    64
#define HH    256
#define WW    256
#define KH    5
#define KW    3
#define KTAP  15          // KH*KW
#define OFFC  30          // 2*KTAP
#define PLANE (HH * WW)   // 65536

typedef __attribute__((ext_vector_type(16))) _Float16 v16h;
typedef __attribute__((ext_vector_type(8)))  _Float16 v8h;
typedef __attribute__((ext_vector_type(8)))  float    v8f;

// ---------------------------------------------------------------------------
// Kernel 1: dual-branch 3x3 conv (offset generators) as implicit GEMM on WMMA
//   M = pixels (128 per block strip), N = 64 (= 30 off1 | pad | 30 off2 | pad),
//   K = 576 decomposed as 9 taps x 64 channels (2 WMMA k-steps of 32 per tap).
//   A-stage is transposed ([col][c], stride 72) so each A fragment is two
//   16-byte ds_load_b128 with a perfect 4-bank tiling across the wave.
// ---------------------------------------------------------------------------
__global__ __launch_bounds__(256) void offsets_conv_wmma(
    const float* __restrict__ x,
    const float* __restrict__ w1, const float* __restrict__ b1,
    const float* __restrict__ w2, const float* __restrict__ b2,
    float* __restrict__ off1, float* __restrict__ off2)
{
    __shared__ _Float16 xrowT[132][72];  // [col][c]: cols xbase-1..xbase+128, padded stride
    __shared__ _Float16 wtap[64][72];    // one tap's weights, N x C (padded rows)

    const int tid  = threadIdx.x;
    const int lane = tid & 31;
    const int wave = tid >> 5;           // 8 waves; wave w owns pixels [16w,16w+16)
    const int b     = blockIdx.z;
    const int y     = blockIdx.y;
    const int xbase = blockIdx.x * 128;

    const int lhalf = lane & 16;         // 0 / 16
    const int l15   = lane & 15;
    const int colA  = wave * 16 + l15;   // this lane's pixel (A-matrix row), 0..127

    v8f acc[4];
    #pragma unroll
    for (int t = 0; t < 4; ++t)
        #pragma unroll
        for (int v = 0; v < 8; ++v) acc[t][v] = 0.0f;

    for (int i = 0; i < 3; ++i) {
        // ---- speculative prefetch of next input row (overlaps with WMMA work) ----
        if (i < 2) {
            const int ny = y + i;        // gy of iteration i+1
            if (ny >= 0 && ny < HH) {
                for (int idx = tid * 32; idx < CH * 130; idx += 256 * 32) {
                    const int c  = idx / 130;
                    const int gx = min(max(xbase + (idx % 130) - 1, 0), WW - 1);
                    __builtin_prefetch(&x[((b * CH + c) * HH + ny) * WW + gx], 0, 0);
                }
            }
        }
        __syncthreads();
        // ---- stage input row gy = y + i - 1 (zero-padded), f32 -> f16, transposed ----
        const int gy = y + i - 1;
        for (int idx = tid; idx < CH * 130; idx += 256) {
            const int col = idx % 130;
            const int c   = idx / 130;
            const int gx  = xbase + col - 1;
            float v = 0.0f;
            if (gy >= 0 && gy < HH && gx >= 0 && gx < WW)
                v = x[((b * CH + c) * HH + gy) * WW + gx];
            xrowT[col][c] = (_Float16)v;
        }
        for (int j = 0; j < 3; ++j) {
            __syncthreads();
            // ---- stage tap (i,j) weights: rows 0..29 = branch1, 32..61 = branch2 ----
            for (int idx = tid; idx < 64 * 64; idx += 256) {
                const int n = idx >> 6, c = idx & 63;
                float wv = 0.0f;
                if (n < OFFC)                 wv = w1[((n * CH + c) * 3 + i) * 3 + j];
                else if (n >= 32 && n < 62)   wv = w2[(((n - 32) * CH + c) * 3 + i) * 3 + j];
                wtap[n][c] = (_Float16)wv;
            }
            __syncthreads();
            #pragma unroll
            for (int s = 0; s < 2; ++s) {
                // A fragment: 16-bit A 16x32 layout. Per lane: c-runs
                // [base, base+8) and [base+16, base+24), base = s*32 + 8*(lane>=16).
                const _Float16* ap = &xrowT[colA + j][s * 32 + (lhalf >> 1)];
                const v8h alo = *(const v8h*)(ap);
                const v8h ahi = *(const v8h*)(ap + 16);
                const v16h afrag = __builtin_shufflevector(
                    alo, ahi, 0, 1, 2, 3, 4, 5, 6, 7, 8, 9, 10, 11, 12, 13, 14, 15);
                #pragma unroll
                for (int t = 0; t < 4; ++t) {
                    // B fragment: 16-bit B 32x16 layout, k-runs [base, base+8) and [base+16, base+24), same as A.
                    const _Float16* bp = &wtap[t * 16 + l15][s * 32 + (lhalf >> 1)];
                    const v8h blo = *(const v8h*)(bp);
                    const v8h bhi = *(const v8h*)(bp + 16);
                    const v16h bfrag = __builtin_shufflevector(
                        blo, bhi, 0, 1, 2, 3, 4, 5, 6, 7, 8, 9, 10, 11, 12, 13, 14, 15);
                    acc[t] = __builtin_amdgcn_wmma_f32_16x16x32_f16(
                        false, afrag, false, bfrag, (short)0, acc[t], false, false);
                }
            }
        }
    }

    // ---- epilogue: D layout m = v + 8*(lane>=16), n = lane&15 ----
    const int mbase = wave * 16 + (lhalf >> 1);
    #pragma unroll
    for (int t = 0; t < 4; ++t) {
        const int n = t * 16 + l15;
        const float* bb;
        float*       ob;
        int          nc;
        if (n < OFFC)               { nc = n;      bb = b1; ob = off1; }
        else if (n >= 32 && n < 62) { nc = n - 32; bb = b2; ob = off2; }
        else continue;
        #pragma unroll
        for (int v = 0; v < 8; ++v) {
            const int px = xbase + mbase + v;
            ob[((b * OFFC + nc) * HH + y) * WW + px] = acc[t][v] + bb[nc];
        }
    }
}

// ---------------------------------------------------------------------------
// Kernel 2: depthwise deformable 5x3 conv + per-(b,c) GAP partial sums.
//   Block = 16x16 pixel tile. Bilinear params computed once per (pixel, tap)
//   into LDS, reused across all 64 channels (8-channel register chunks).
//   x is L2-resident (64 MB < 192 MB) so the 4 gathers/tap hit L2.
// ---------------------------------------------------------------------------
__global__ __launch_bounds__(256) void deform_dw(
    const float* __restrict__ xin, const float* __restrict__ off,
    const float* __restrict__ wk,  const float* __restrict__ bias,
    float* __restrict__ out, float* __restrict__ gsum)
{
    __shared__ int   sy0[KTAP * 256];
    __shared__ int   sx0[KTAP * 256];
    __shared__ float sly[KTAP * 256];
    __shared__ float slx[KTAP * 256];
    __shared__ float ldsg[CH];

    const int tid  = threadIdx.x;
    const int lane = tid & 31;
    const int b  = blockIdx.z;
    const int y  = blockIdx.y * 16 + (tid >> 4);
    const int xx = blockIdx.x * 16 + (tid & 15);

    // stage 1: bilinear parameters per tap (private LDS slots, SoA, no conflicts)
    #pragma unroll
    for (int k = 0; k < KTAP; ++k) {
        const int i = k / KW, j = k % KW;
        const float dy = off[((b * OFFC + 2 * k)     * HH + y) * WW + xx];
        const float dx = off[((b * OFFC + 2 * k + 1) * HH + y) * WW + xx];
        const float py = (float)(y - 2 + i) + dy;
        const float px = (float)(xx - 1 + j) + dx;
        const float fy = floorf(py), fx = floorf(px);
        sy0[k * 256 + tid] = (int)fy;
        sx0[k * 256 + tid] = (int)fx;
        sly[k * 256 + tid] = py - fy;
        slx[k * 256 + tid] = px - fx;
    }
    if (tid < CH) ldsg[tid] = 0.0f;
    __syncthreads();

    // stage 2: channel chunks of 8
    for (int cb = 0; cb < CH; cb += 8) {
        float acc[8];
        #pragma unroll
        for (int cc = 0; cc < 8; ++cc) acc[cc] = bias[cb + cc];

        #pragma unroll
        for (int k = 0; k < KTAP; ++k) {
            const int   y0 = sy0[k * 256 + tid];
            const int   x0 = sx0[k * 256 + tid];
            const float ly = sly[k * 256 + tid];
            const float lx = slx[k * 256 + tid];
            const bool vy0 = (y0 >= 0)     & (y0 < HH);
            const bool vy1 = (y0 + 1 >= 0) & (y0 + 1 < HH);
            const bool vx0 = (x0 >= 0)     & (x0 < WW);
            const bool vx1 = (x0 + 1 >= 0) & (x0 + 1 < WW);
            const int yc0 = min(max(y0, 0), HH - 1),  yc1 = min(max(y0 + 1, 0), HH - 1);
            const int xc0 = min(max(x0, 0), WW - 1),  xc1 = min(max(x0 + 1, 0), WW - 1);
            const float w00 = (1.f - ly) * (1.f - lx) * (float)(vy0 & vx0);
            const float w01 = (1.f - ly) * lx         * (float)(vy0 & vx1);
            const float w10 = ly * (1.f - lx)         * (float)(vy1 & vx0);
            const float w11 = ly * lx                 * (float)(vy1 & vx1);
            const int i00 = yc0 * WW + xc0, i01 = yc0 * WW + xc1;
            const int i10 = yc1 * WW + xc0, i11 = yc1 * WW + xc1;
            #pragma unroll
            for (int cc = 0; cc < 8; ++cc) {
                const float* plane = xin + (size_t)(b * CH + cb + cc) * PLANE;
                const float s = w00 * plane[i00] + w01 * plane[i01]
                              + w10 * plane[i10] + w11 * plane[i11];
                acc[cc] += wk[(cb + cc) * KTAP + k] * s;
            }
        }
        #pragma unroll
        for (int cc = 0; cc < 8; ++cc) {
            out[((b * CH + cb + cc) * HH + y) * WW + xx] = acc[cc];
            // wave32 reduce -> LDS accumulator (GAP partial)
            float v = acc[cc];
            #pragma unroll
            for (int o = 16; o > 0; o >>= 1) v += __shfl_xor(v, o, 32);
            if (lane == 0) atomicAdd(&ldsg[cb + cc], v);
        }
    }
    __syncthreads();
    if (tid < CH) atomicAdd(&gsum[b * CH + tid], ldsg[tid]);
}

// ---------------------------------------------------------------------------
// Kernel 3: zero the GAP accumulators
// ---------------------------------------------------------------------------
__global__ void zero_g(float* __restrict__ g)
{
    g[threadIdx.x] = 0.0f;   // 512 threads: g1(256) ++ g2(256)
}

// ---------------------------------------------------------------------------
// Kernel 4: channel attention (softmax over 2 branches) + weighted sum.
//   out2 lives in d_out; rewritten in place. float4 vectorized.
// ---------------------------------------------------------------------------
__global__ __launch_bounds__(256) void combine(
    const float* __restrict__ out1, float* __restrict__ out,
    const float* __restrict__ g1, const float* __restrict__ g2)
{
    const int idx4 = blockIdx.x * 256 + threadIdx.x;   // float4 index
    const int bc   = (idx4 * 4) >> 16;                 // (b*CH + c), PLANE = 65536
    const float m1 = g1[bc] * (1.0f / PLANE);
    const float m2 = g2[bc] * (1.0f / PLANE);
    const float mm = fmaxf(m1, m2);
    const float e1 = __expf(m1 - mm), e2 = __expf(m2 - mm);
    const float inv = 1.0f / (e1 + e2);
    const float a1 = e1 * inv, a2 = e2 * inv;
    const float4 o1 = ((const float4*)out1)[idx4];
    const float4 o2 = ((const float4*)out)[idx4];
    float4 r;
    r.x = a1 * o1.x + a2 * o2.x;
    r.y = a1 * o1.y + a2 * o2.y;
    r.z = a1 * o1.z + a2 * o2.z;
    r.w = a1 * o1.w + a2 * o2.w;
    ((float4*)out)[idx4] = r;
}

// ---------------------------------------------------------------------------
// Launch
// ---------------------------------------------------------------------------
extern "C" void kernel_launch(void* const* d_in, const int* in_sizes, int n_in,
                              void* d_out, int out_size, void* d_ws, size_t ws_size,
                              hipStream_t stream)
{
    const float* x      = (const float*)d_in[0];
    const float* w_off1 = (const float*)d_in[1];
    const float* b_off1 = (const float*)d_in[2];
    const float* w_dcn1 = (const float*)d_in[3];
    const float* b_dcn1 = (const float*)d_in[4];
    const float* w_off2 = (const float*)d_in[5];
    const float* b_off2 = (const float*)d_in[6];
    const float* w_dcn2 = (const float*)d_in[7];
    const float* b_dcn2 = (const float*)d_in[8];
    float* out = (float*)d_out;

    // workspace: off1 | off2 | out1 | g1 | g2
    float* off1 = (float*)d_ws;
    float* off2 = off1 + (size_t)BATCH * OFFC * PLANE;
    float* out1 = off2 + (size_t)BATCH * OFFC * PLANE;
    float* g1   = out1 + (size_t)BATCH * CH * PLANE;
    float* g2   = g1 + BATCH * CH;

    zero_g<<<1, 2 * BATCH * CH, 0, stream>>>(g1);

    offsets_conv_wmma<<<dim3(WW / 128, HH, BATCH), 256, 0, stream>>>(
        x, w_off1, b_off1, w_off2, b_off2, off1, off2);

    deform_dw<<<dim3(WW / 16, HH / 16, BATCH), 256, 0, stream>>>(
        x, off1, w_dcn1, b_dcn1, out1, g1);

    deform_dw<<<dim3(WW / 16, HH / 16, BATCH), 256, 0, stream>>>(
        out1, off2, w_dcn2, b_dcn2, out, g2);   // out2 -> d_out (temp)

    combine<<<dim3((BATCH * CH * PLANE / 4) / 256), 256, 0, stream>>>(
        out1, out, g1, g2);
}
